// PackedAttention_73194832658652
// MI455X (gfx1250) — matrix operand order repaired and checked by reading, not compiled, not run on hardware
//
#include <hip/hip_runtime.h>

// ---------------------------------------------------------------------------
// Packed varlen attention (block-diagonal, non-causal) for MI455X / gfx1250.
//
// H=16, T=4096, D=64, fp32 in/out, NSEG=8 segments (cu_seqlens: 9 int32).
//
// Flash attention with WMMA f16->f32:
//   S^T = K * Q^T   (A = K tile 16x32, B = Q^T 32x16)  -> C layout of S^T is
//   already the A-fragment layout of P for  O = P * V  (no shuffles needed).
// Pre-pass converts K -> f16 (row major) and V -> f16 transposed vT[h][d][t]
// in d_ws so every hot-loop fragment is a contiguous 16B uint4 (b128) load.
// Workspace: 2 * H*T*D * 2B = 16 MB (fits in 192MB L2 -> tile re-reads across
// the 256 q-tile waves per head are L2 hits; HBM sees ~50MB once).
//
// 64 keys per iteration (16 WMMAs/iter) to amortize softmax bookkeeping;
// v_cvt_pk_rtz_f16_f32 packs the P fragment (2 keys/op). Wave-uniform
// interior fast path + ballot-gated O-rescale.
// ---------------------------------------------------------------------------

namespace {
constexpr int Hn    = 16;
constexpr int Tn    = 4096;
constexpr int Dn    = 64;
constexpr int NSEG  = 8;
constexpr int WAVES = 8;   // waves per block (8 * 32 = 256 threads)
}

typedef __attribute__((ext_vector_type(16))) _Float16 v16h;
typedef __attribute__((ext_vector_type(8)))  float    v8f;

union F16Frag {                        // 16 halves = one WMMA f16 A/B fragment
  uint4    u[2];
  unsigned u32[8];
  v16h     h;
};
union H4 { _Float16 h[4]; uint2 u; };
union HB { _Float16 h; unsigned short s; };

__device__ __forceinline__ float bperm_f(int srcLane, float v) {
  // wave32: only index bits [6:2] are used; srcLane in [0,31]
  return __int_as_float(__builtin_amdgcn_ds_bpermute(srcLane << 2, __float_as_int(v)));
}
__device__ __forceinline__ unsigned pk_f16(float a, float b) {
  // v_cvt_pk_rtz_f16_f32; builtin returns __fp16 ext_vector(2) (4 bytes)
  return __builtin_bit_cast(unsigned, __builtin_amdgcn_cvt_pkrtz(a, b));
}

// ---------------------------------------------------------------------------
// Pre-pass 1: K fp32 -> f16, same layout kf[((h*T)+t)*D + d]
// ---------------------------------------------------------------------------
__global__ __launch_bounds__(256) void cvt_k_f16(const float* __restrict__ k,
                                                 unsigned short* __restrict__ kf) {
  int i = (blockIdx.x * 256 + threadIdx.x) * 4;
  float4 x = *(const float4*)(k + i);
  H4 o;
  o.h[0] = (_Float16)x.x; o.h[1] = (_Float16)x.y;
  o.h[2] = (_Float16)x.z; o.h[3] = (_Float16)x.w;
  *(uint2*)(kf + i) = o.u;
}

// ---------------------------------------------------------------------------
// Pre-pass 2: V fp32 [h][t][d] -> f16 transposed vT[h][d][t] (LDS-tiled, 64x64)
// ---------------------------------------------------------------------------
__global__ __launch_bounds__(256) void transpose_v_f16(const float* __restrict__ v,
                                                       unsigned short* __restrict__ vT) {
  __shared__ unsigned short tile[64][65];   // +1 pad: no bank conflicts
  const int h  = blockIdx.x >> 6;           // T/64 = 64 tiles per head
  const int t0 = (blockIdx.x & 63) * 64;
  const int tid = threadIdx.x;
#pragma unroll
  for (int i = 0; i < 16; ++i) {
    int e = tid + i * 256;                  // 0..4095
    int r = e >> 6, c = e & 63;             // token row r, dim col c (c fastest: coalesced)
    HB b; b.h = (_Float16)v[((h * Tn) + t0 + r) * Dn + c];
    tile[c][r] = b.s;                       // store transposed
  }
  __syncthreads();
#pragma unroll
  for (int i = 0; i < 16; ++i) {
    int e = tid + i * 256;
    int d = e >> 6, tt = e & 63;            // tt fastest: coalesced writes
    vT[((h * Dn) + d) * Tn + t0 + tt] = tile[d][tt];
  }
}

// ---------------------------------------------------------------------------
// Main kernel: one wave32 per (head, 16-query tile), 64 keys per iteration.
// ---------------------------------------------------------------------------
__global__ __launch_bounds__(256) void attn_varlen_wmma(
    const float* __restrict__ q,
    const unsigned short* __restrict__ kf,   // f16 K, row major
    const unsigned short* __restrict__ vT,   // f16 V^T: [h][d][t]
    const int* __restrict__ cu,              // cu_seqlens, NSEG+1 ints
    float* __restrict__ out) {

  const int wave  = (blockIdx.x * WAVES) + (threadIdx.x >> 5);
  const int lane  = threadIdx.x & 31;
  const int h     = wave >> 8;               // 256 q-tiles per head
  const int qbase = (wave & 255) << 4;
  const int ln15  = lane & 15;
  const int qHalf = lane >> 4;               // 0: lanes 0-15, 1: lanes 16-31
  const int qh8   = qHalf * 8;
  const int qrow  = qbase + ln15;            // this lane's query column in S^T

  // --- segment [s0,s1) containing qrow (searchsorted(cu[1:], q, 'right')) ---
  int s0 = 0, s1 = Tn;
#pragma unroll
  for (int s = 1; s <= NSEG; ++s) {
    int c = cu[s];
    if (c <= qrow)      s0 = c;              // boundaries sorted ascending
    else if (c < s1)    s1 = c;              // first boundary > qrow
  }
  // wave-uniform bounds: rows sorted -> s0 non-decreasing, s1 non-decreasing
  const int kLo    = __builtin_amdgcn_readlane(s0, 0);   // union start
  const int kHi    = __builtin_amdgcn_readlane(s1, 15);  // union end
  const int kLoMax = __builtin_amdgcn_readlane(s0, 15);  // intersection start
  const int kHiMin = __builtin_amdgcn_readlane(s1, 0);   // intersection end

  // --- Q^T B-fragments (loaded once): lane = query col, halves = d rows ----
  // fold softmax scale and log2(e) so we can use v_exp_f32 (base 2)
  const float qscale = 0.125f * 1.44269504088896340736f;
  F16Frag qf[2];
  {
    const float* qrp = q + ((size_t)(h * Tn + qrow)) * Dn;
#pragma unroll
    for (int g = 0; g < 2; ++g) {
      int db = g * 32 + qHalf * 16;          // B rows d = db .. db+15
#pragma unroll
      for (int c4 = 0; c4 < 4; ++c4) {
        float4 f = *(const float4*)(qrp + db + c4 * 4);
        qf[g].u32[c4 * 2 + 0] = pk_f16(f.x * qscale, f.y * qscale);
        qf[g].u32[c4 * 2 + 1] = pk_f16(f.z * qscale, f.w * qscale);
      }
    }
  }

  v8f oacc[4] = {v8f{}, v8f{}, v8f{}, v8f{}};   // O tiles, d cols g*16 + ln15
  float m = -__builtin_inff();
  float l = 0.0f;

  const int kk0   = kLo & ~63;
  const int kkEnd = (kHi + 63) & ~63;           // <= T (T % 64 == 0)

  for (int kk = kk0; kk < kkEnd; kk += 64) {
    // ---- S^T = K * Q^T for key subtiles kk+{0,16,32,48}..+15 --------------
    v8f cs[4] = {v8f{}, v8f{}, v8f{}, v8f{}};
    const unsigned short* krow = kf + ((size_t)(h * Tn + kk + ln15)) * Dn;
#pragma unroll
    for (int g = 0; g < 2; ++g) {
      const int db = g * 32;
#pragma unroll
      for (int t = 0; t < 4; ++t) {
        const unsigned short* kr = krow + (size_t)(t * 16) * Dn;
        F16Frag a;
        a.u[0] = *(const uint4*)(kr + db + qh8 * 2);        // halves 0-7 : d = db+16*qHalf..
        a.u[1] = *(const uint4*)(kr + db + qh8 * 2 + 8);    // halves 8-15: d = db+16*qHalf+8..
        cs[t] = __builtin_amdgcn_wmma_f32_16x16x32_f16(false, a.h, false, qf[g].h,
                                                       (short)0, cs[t], false, false);
      }
    }

    // ---- online softmax (per lane: 32 keys of one query) ------------------
    float mNew, alpha, lLoc;
    F16Frag pfA, pfB;                    // P A-fragments: keys kk..+31, kk+32..+63

    // wave-uniform: is this key tile fully valid for all 16 query rows?
    if (kk >= kLoMax && kk + 64 <= kHiMin) {
      // -------- fast path: no masking, no exp guards ----------------------
      float tmax = -__builtin_inff();
#pragma unroll
      for (int t = 0; t < 4; ++t)
#pragma unroll
        for (int vv = 0; vv < 8; ++vv) tmax = fmaxf(tmax, cs[t][vv]);
      tmax = fmaxf(tmax, bperm_f((lane ^ 16) & 31, tmax));   // pair combine
      mNew  = fmaxf(m, tmax);                                // finite
      alpha = __builtin_amdgcn_exp2f(m - mNew);              // m=-inf -> 0, ok
      lLoc = 0.0f;
#pragma unroll
      for (int t = 0; t < 4; ++t) {
        F16Frag* pf = (t < 2) ? &pfA : &pfB;
        const int hb = (t & 1) * 4;      // subtile even -> halves 0-7, odd -> 8-15
#pragma unroll
        for (int vv = 0; vv < 8; vv += 2) {
          float p0 = __builtin_amdgcn_exp2f(cs[t][vv]     - mNew);
          float p1 = __builtin_amdgcn_exp2f(cs[t][vv + 1] - mNew);
          lLoc += p0 + p1;
          pf->u32[hb + vv / 2] = pk_f16(p0, p1);
        }
      }
    } else {
      // -------- slow path: boundary tile, mask + guarded exp --------------
      float tmax = -__builtin_inff();
#pragma unroll
      for (int t = 0; t < 4; ++t)
#pragma unroll
        for (int vv = 0; vv < 8; ++vv) {
          int key = kk + t * 16 + qh8 + vv;  // C layout: vgpr vv <-> key row vv+8*qHalf
          float x = (key >= s0 && key < s1) ? cs[t][vv] : -__builtin_inff();
          cs[t][vv] = x;
          tmax = fmaxf(tmax, x);
        }
      tmax = fmaxf(tmax, bperm_f((lane ^ 16) & 31, tmax));
      mNew  = fmaxf(m, tmax);
      alpha = (mNew > -1e37f) ? __builtin_amdgcn_exp2f(m - mNew) : 0.0f;
      lLoc = 0.0f;
#pragma unroll
      for (int t = 0; t < 4; ++t) {
        F16Frag* pf = (t < 2) ? &pfA : &pfB;
        const int hb = (t & 1) * 4;
#pragma unroll
        for (int vv = 0; vv < 8; vv += 2) {
          float p0 = (cs[t][vv]     > -1e37f) ? __builtin_amdgcn_exp2f(cs[t][vv]     - mNew) : 0.0f;
          float p1 = (cs[t][vv + 1] > -1e37f) ? __builtin_amdgcn_exp2f(cs[t][vv + 1] - mNew) : 0.0f;
          lLoc += p0 + p1;
          pf->u32[hb + vv / 2] = pk_f16(p0, p1);
        }
      }
    }

    const float lPair = bperm_f((lane ^ 16) & 31, lLoc);

    // ---- O rescale, skipped (uniformly) when no row's max moved -----------
    if (__ballot(mNew > m) != 0ull) {
      float ar[8];
#pragma unroll
      for (int vv = 0; vv < 8; ++vv) ar[vv] = bperm_f(qh8 + vv, alpha);
#pragma unroll
      for (int g = 0; g < 4; ++g)
#pragma unroll
        for (int vv = 0; vv < 8; ++vv) oacc[g][vv] *= ar[vv];
      l = l * alpha + lLoc + lPair;
      m = mNew;
    } else {
      l = l + lLoc + lPair;              // alpha == 1 for every row
    }

    // ---- O += P(16x64) * V(64x16) for the four 16-wide d column groups ----
#pragma unroll
    for (int g = 0; g < 4; ++g) {
      const int dcol = g * 16 + ln15;    // B col = lane%16, rows k contiguous in vT
      const unsigned short* vrp = vT + ((size_t)(h * Dn + dcol)) * Tn + kk + qh8;
      F16Frag vb0, vb1;
      vb0.u[0] = *(const uint4*)(vrp);
      vb0.u[1] = *(const uint4*)(vrp + 16);
      vb1.u[0] = *(const uint4*)(vrp + 32);
      vb1.u[1] = *(const uint4*)(vrp + 48);
      oacc[g] = __builtin_amdgcn_wmma_f32_16x16x32_f16(false, pfA.h, false, vb0.h,
                                                       (short)0, oacc[g], false, false);
      oacc[g] = __builtin_amdgcn_wmma_f32_16x16x32_f16(false, pfB.h, false, vb1.h,
                                                       (short)0, oacc[g], false, false);
    }
  }

  // ---- normalize by l (broadcast per O row) and store -----------------------
  float inv[8];
#pragma unroll
  for (int vv = 0; vv < 8; ++vv) inv[vv] = 1.0f / bperm_f(qh8 + vv, l);

#pragma unroll
  for (int g = 0; g < 4; ++g) {
#pragma unroll
    for (int vv = 0; vv < 8; ++vv) {
      const int qr = qbase + qh8 + vv;    // C layout: vgpr vv -> row vv + 8*qHalf
      out[((size_t)(h * Tn + qr)) * Dn + g * 16 + ln15] = oacc[g][vv] * inv[vv];
    }
  }
}

// ---------------------------------------------------------------------------
extern "C" void kernel_launch(void* const* d_in, const int* in_sizes, int n_in,
                              void* d_out, int out_size, void* d_ws, size_t ws_size,
                              hipStream_t stream) {
  (void)in_sizes; (void)n_in; (void)out_size; (void)ws_size;

  const float* q  = (const float*)d_in[0];
  const float* k  = (const float*)d_in[1];
  const float* v  = (const float*)d_in[2];
  const int*   cu = (const int*)d_in[3];
  float* out = (float*)d_out;

  // workspace: f16 K copy (8 MB) + f16 V^T (8 MB)
  unsigned short* kf = (unsigned short*)d_ws;
  unsigned short* vT = kf + (size_t)Hn * Tn * Dn;

  const int nElem = Hn * Tn * Dn;                       // 4,194,304
  cvt_k_f16<<<nElem / (256 * 4), 256, 0, stream>>>(k, kf);
  transpose_v_f16<<<Hn * (Tn / 64), 256, 0, stream>>>(v, vT);

  const int nTiles = Hn * (Tn / 16);                    // 4096 waves
  attn_varlen_wmma<<<nTiles / WAVES, WAVES * 32, 0, stream>>>(q, kf, vT, cu, out);
}
